// RelativePositionalMultiHeadAttn2_88192858456501
// MI455X (gfx1250) — compile-verified, matrix operand order checked by reading, not run on hardware
//
#include <hip/hip_runtime.h>

// ---------------------------------------------------------------------------
// Fused Swin windowed MHA for gfx1250 (MI455X), wave32 + v_wmma_f32_16x16x32_f16
// ---------------------------------------------------------------------------

typedef __attribute__((ext_vector_type(16))) _Float16 v16h;
typedef __attribute__((ext_vector_type(8)))  float    v8f;

static constexpr int SEQ  = 64;    // tokens per window (P)
static constexpr int CH   = 384;   // channels (D)
static constexpr int NH   = 12;    // heads
static constexpr int HD   = 32;    // head dim
static constexpr int C3   = 1152;  // 3*D
static constexpr float QK_SCALE = 0.17677669529663687f; // 32^-0.5

// LDS strides (in _Float16 elements) with padding to dodge bank conflicts
static constexpr int QS_STRIDE = CH + 8;   // 392
static constexpr int PS_STRIDE = SEQ + 4;  // 68
static constexpr int VT_STRIDE = SEQ + 4;  // 68

// LDS byte offsets (all 16B aligned)
static constexpr int REL_OFF = 0;                                   // 225*12 f32 = 10800 -> pad
static constexpr int QS_OFF  = 10816;
static constexpr int KS_OFF  = QS_OFF + SEQ * QS_STRIDE * 2;        // 60992
static constexpr int VT_OFF  = KS_OFF + SEQ * QS_STRIDE * 2;        // 111168
static constexpr int OS_OFF  = VT_OFF + NH * HD * VT_STRIDE * 2;    // 163392
static constexpr int PS_OFF  = OS_OFF + SEQ * QS_STRIDE * 2;        // 213568
static constexpr int XS_OFF  = PS_OFF;                              // xs aliases ps (phase-1 only)
static constexpr int LDS_BYTES = PS_OFF + 8 * SEQ * PS_STRIDE * 2;  // 283200 (< 320KB WGP LDS)

__device__ __forceinline__ v8f wmma_f16(v16h a, v16h b, v8f c) {
  return __builtin_amdgcn_wmma_f32_16x16x32_f16(
      /*neg_a=*/false, a, /*neg_b=*/false, b,
      /*c_mod=*/(short)0, c, /*reuse_a=*/false, /*reuse_b=*/false);
}

// Load a 16x32 WMMA fragment (A-layout; B uses identical layout with N<->M)
// from a row-major array whose "rows" hold the K dimension contiguously.
// ISA layout: lanes 0-15 -> K {0..7,16..23}; lanes 16-31 -> K {8..15,24..31}.
template <typename T>
__device__ __forceinline__ v16h load_frag(const T* base, int row0, int stride, int kbase) {
  const int lane = threadIdx.x & 31;
  const int hh   = (lane >> 4) * 8;
  const int rr   = lane & 15;
  const T* p = base + (size_t)(row0 + rr) * stride + kbase + hh;
  v16h f;
#pragma unroll
  for (int v = 0; v < 8; ++v) {
    const int kk = (v >> 2) * 16 + (v & 3) * 2;   // 0,2,4,6,16,18,20,22
    f[2 * v]     = (_Float16)p[kk];
    f[2 * v + 1] = (_Float16)p[kk + 1];
  }
  return f;
}

// ---------------------------------------------------------------------------
// Prep: transpose + f32->f16 convert weights so B-fragment loads are
// contiguous per lane. wqkvT: [1152][384], wmT: [384][384].
// ---------------------------------------------------------------------------
__global__ void prep_weights(const float* __restrict__ Wqkv,
                             const float* __restrict__ Wm,
                             _Float16* __restrict__ wqkvT,
                             _Float16* __restrict__ wmT) {
  const int i = blockIdx.x * blockDim.x + threadIdx.x;
  const int nq = C3 * CH;
  if (i < nq) {
    const int n = i / CH, k = i % CH;
    wqkvT[i] = (_Float16)Wqkv[(size_t)k * C3 + n];
  } else if (i < nq + CH * CH) {
    const int j = i - nq;
    const int n = j / CH, k = j % CH;
    wmT[j] = (_Float16)Wm[(size_t)k * CH + n];
  }
}

// ---------------------------------------------------------------------------
// One block per window (B*G = 2048 blocks), 256 threads = 8 wave32 waves.
// ---------------------------------------------------------------------------
__global__ __launch_bounds__(256) void
swin_window_attn(const float* __restrict__ x,
                 const float* __restrict__ bqkv,
                 const float* __restrict__ bm,
                 const float* __restrict__ rel_table,
                 const _Float16* __restrict__ wqkvT,
                 const _Float16* __restrict__ wmT,
                 float* __restrict__ out) {
  extern __shared__ char smem[];
  float*    rel = (float*)   (smem + REL_OFF);
  _Float16* qs  = (_Float16*)(smem + QS_OFF);
  _Float16* ks  = (_Float16*)(smem + KS_OFF);
  _Float16* vt  = (_Float16*)(smem + VT_OFF);
  _Float16* os  = (_Float16*)(smem + OS_OFF);
  _Float16* psA = (_Float16*)(smem + PS_OFF);
  _Float16* xs  = (_Float16*)(smem + XS_OFF);

  const int win  = blockIdx.x;
  const int tid  = threadIdx.x;
  const int lane = tid & 31;
  const int wave = tid >> 5;
  const int hlf  = lane >> 4;   // which 16-lane half
  const int l15  = lane & 15;

  // stage relative-position table (225 x 12 f32)
  for (int i = tid; i < 225 * NH; i += 256) rel[i] = rel_table[i];

  // stage x window [64 x 384] as f16
  const float* xg = x + (size_t)win * (SEQ * CH);
  for (int i = tid; i < SEQ * CH; i += 256) {
    const int r = i / CH, c = i % CH;
    xs[r * QS_STRIDE + c] = (_Float16)xg[i];
  }
  __syncthreads();

  // ---------------- Phase 1: qkv = x @ Wqkv + bqkv ----------------
  for (int t = wave; t < 4 * 72; t += 8) {
    const int mt = t / 72, nt = t % 72;
    v8f acc = {};
#pragma unroll 4
    for (int k0 = 0; k0 < 12; ++k0) {
      v16h a  = load_frag(xs,    mt * 16, QS_STRIDE, k0 * 32);
      v16h bf = load_frag(wqkvT, nt * 16, CH,        k0 * 32);
      acc = wmma_f16(a, bf, acc);
    }
    const int   ncol = nt * 16 + l15;
    const float bb   = bqkv[ncol];
    const int   rb   = mt * 16 + hlf * 8;
    if (ncol < CH) {                         // Q (scaled)
#pragma unroll
      for (int r = 0; r < 8; ++r)
        qs[(rb + r) * QS_STRIDE + ncol] = (_Float16)((acc[r] + bb) * QK_SCALE);
    } else if (ncol < 2 * CH) {              // K
#pragma unroll
      for (int r = 0; r < 8; ++r)
        ks[(rb + r) * QS_STRIDE + (ncol - CH)] = (_Float16)(acc[r] + bb);
    } else {                                 // V, stored transposed per head
      const int h = (ncol - 2 * CH) >> 5, d = (ncol - 2 * CH) & 31;
#pragma unroll
      for (int r = 0; r < 8; ++r)
        vt[(h * HD + d) * VT_STRIDE + rb + r] = (_Float16)(acc[r] + bb);
    }
  }
  __syncthreads();

  // ---------------- Phase 2: per-head attention ----------------
  _Float16* ps = psA + wave * (SEQ * PS_STRIDE);
  for (int h = wave; h < NH; h += 8) {
    // S = Q Kt + bias, softmax rows, probs -> ps (f16)
    for (int ti = 0; ti < 4; ++ti) {
      v16h a = load_frag(qs, ti * 16, QS_STRIDE, h * HD);
      v8f s[4];
#pragma unroll
      for (int tj = 0; tj < 4; ++tj) {
        v16h bf = load_frag(ks, tj * 16, QS_STRIDE, h * HD);
        v8f z = {};
        s[tj] = wmma_f16(a, bf, z);
      }
#pragma unroll
      for (int r = 0; r < 8; ++r) {
        const int i  = ti * 16 + hlf * 8 + r;
        const int ih = i >> 3, iw = i & 7;
        float m = -1e30f;
#pragma unroll
        for (int tj = 0; tj < 4; ++tj) {
          const int j   = tj * 16 + l15;
          const int idx = ((ih - (j >> 3) + 7) * 15 + (iw - (j & 7) + 7)) * NH + h;
          const float v = s[tj][r] + rel[idx];
          s[tj][r] = v;
          m = fmaxf(m, v);
        }
        for (int d = 1; d < 16; d <<= 1) m = fmaxf(m, __shfl_xor(m, d, 32));
        float sum = 0.f;
#pragma unroll
        for (int tj = 0; tj < 4; ++tj) {
          const float e = __expf(s[tj][r] - m);
          s[tj][r] = e;
          sum += e;
        }
        for (int d = 1; d < 16; d <<= 1) sum += __shfl_xor(sum, d, 32);
        const float inv = 1.0f / sum;
#pragma unroll
        for (int tj = 0; tj < 4; ++tj)
          ps[i * PS_STRIDE + tj * 16 + l15] = (_Float16)(s[tj][r] * inv);
      }
    }
    // O_h = P @ V_h  (K = 64 -> 2 WMMA k-steps)
    const _Float16* vth = vt + h * HD * VT_STRIDE;
    for (int ti = 0; ti < 4; ++ti)
      for (int dt = 0; dt < 2; ++dt) {
        v8f acc = {};
#pragma unroll
        for (int k0 = 0; k0 < 2; ++k0) {
          v16h a  = load_frag(ps,  ti * 16, PS_STRIDE, k0 * 32);
          v16h bf = load_frag(vth, dt * 16, VT_STRIDE, k0 * 32);
          acc = wmma_f16(a, bf, acc);
        }
#pragma unroll
        for (int r = 0; r < 8; ++r) {
          const int row = ti * 16 + hlf * 8 + r;
          os[row * QS_STRIDE + h * HD + dt * 16 + l15] = (_Float16)acc[r];
        }
      }
  }
  __syncthreads();

  // ---------------- Phase 3: out = O @ Wm + bm ----------------
  float* og = out + (size_t)win * (SEQ * CH);
  for (int t = wave; t < 4 * 24; t += 8) {
    const int mt = t / 24, nt = t % 24;
    v8f acc = {};
#pragma unroll 4
    for (int k0 = 0; k0 < 12; ++k0) {
      v16h a  = load_frag(os,  mt * 16, QS_STRIDE, k0 * 32);
      v16h bf = load_frag(wmT, nt * 16, CH,        k0 * 32);
      acc = wmma_f16(a, bf, acc);
    }
    const int   ncol = nt * 16 + l15;
    const float bb   = bm[ncol];
#pragma unroll
    for (int r = 0; r < 8; ++r) {
      const int row = mt * 16 + hlf * 8 + r;
      og[row * CH + ncol] = acc[r] + bb;
    }
  }
}

// ---------------------------------------------------------------------------
extern "C" void kernel_launch(void* const* d_in, const int* in_sizes, int n_in,
                              void* d_out, int out_size, void* d_ws, size_t ws_size,
                              hipStream_t stream) {
  (void)n_in; (void)out_size; (void)ws_size;
  const float* x         = (const float*)d_in[0];
  const float* Wqkv      = (const float*)d_in[1];
  const float* bqkv      = (const float*)d_in[2];
  const float* Wm        = (const float*)d_in[3];
  const float* bm        = (const float*)d_in[4];
  const float* rel_table = (const float*)d_in[5];
  float* out = (float*)d_out;

  _Float16* wqkvT = (_Float16*)d_ws;
  _Float16* wmT   = (_Float16*)((char*)d_ws + (size_t)C3 * CH * sizeof(_Float16));

  const int nprep = C3 * CH + CH * CH;
  prep_weights<<<(nprep + 255) / 256, 256, 0, stream>>>(Wqkv, Wm, wqkvT, wmT);

  const int nwin = in_sizes[0] / (SEQ * CH);   // B*G = 2048
  swin_window_attn<<<nwin, 256, LDS_BYTES, stream>>>(x, bqkv, bm, rel_table,
                                                     wqkvT, wmT, out);
}
